// TriangleAttention_30090540875872
// MI455X (gfx1250) — matrix operand order repaired and checked by reading, not compiled, not run on hardware
//
#include <hip/hip_runtime.h>
#include <hip/hip_bf16.h>
#include <math.h>

// ---------------- types ----------------
typedef __attribute__((ext_vector_type(16))) _Float16 v16h;
typedef __attribute__((ext_vector_type(8)))  _Float16 v8h;
typedef __attribute__((ext_vector_type(4)))  _Float16 v4h;
typedef __attribute__((ext_vector_type(8)))  float    v8f;
typedef __attribute__((ext_vector_type(4)))  float    v4f;
typedef __attribute__((ext_vector_type(4)))  unsigned int v4u;
typedef __attribute__((ext_vector_type(4)))  int      v4i;
typedef __attribute__((ext_vector_type(8)))  int      v8i;

#define NN   256
#define CZ   128
#define NH   4
#define DH   32
#define INFB 1e9f

#if __has_builtin(__builtin_amdgcn_tensor_load_to_lds)
#define HAVE_TDM 1
#else
#define HAVE_TDM 0
#endif

// ---------------- WMMA helpers (gfx1250 layouts, wave32) ----------------
__device__ __forceinline__ v16h cat8(v8h lo, v8h hi) {
    v16h r;
#pragma unroll
    for (int i = 0; i < 8; ++i) { r[i] = lo[i]; r[i + 8] = hi[i]; }
    return r;
}

// A fragment (16x32 f16, row-major src, leading dim ld in halfs).
// lane<16: row=lane, K = 0..7 & 16..23 ; lane>=16: row=lane-16, K = 8..15 & 24..31
__device__ __forceinline__ v16h load_a(const _Float16* src, int ld, int lane) {
    int row = lane & 15;
    int k0  = (lane < 16) ? 0 : 8;
    const _Float16* p = src + row * ld + k0;
    v8h lo = *(const v8h*)p;
    v8h hi = *(const v8h*)(p + 16);
    return cat8(lo, hi);
}

// B fragment (32x16 f16) from transposed storage Bt[n][k] (row-major, ld in halfs).
// lane<16: col=lane, K = 0..7 & 16..23 ; lane>=16: col=lane-16, K = 8..15 & 24..31
__device__ __forceinline__ v16h load_b(const _Float16* bt, int ld, int lane) {
    int col = lane & 15;
    int kb  = (lane < 16) ? 0 : 8;
    const _Float16* p = bt + col * ld + kb;
    v8h lo = *(const v8h*)p;
    v8h hi = *(const v8h*)(p + 16);
    return cat8(lo, hi);
}

__device__ __forceinline__ v8f wmma_f16(v16h a, v16h b, v8f c) {
    return __builtin_amdgcn_wmma_f32_16x16x32_f16(false, a, false, b, (short)0, c,
                                                  false, false);
}

__device__ __forceinline__ float wave_sum(float v) {
#pragma unroll
    for (int m = 16; m; m >>= 1) v += __shfl_xor(v, m, 32);
    return v;
}
__device__ __forceinline__ float wave_max(float v) {
#pragma unroll
    for (int m = 16; m; m >>= 1) v = fmaxf(v, __shfl_xor(v, m, 32));
    return v;
}

#if HAVE_TDM
__device__ __forceinline__ unsigned lds_addr_of(const void* p) {
    return (unsigned)(unsigned long long)(const __attribute__((address_space(3))) char*)p;
}
#endif

// ---------------- kernel 0: weight convert (fold q scale) ----------------
__global__ __launch_bounds__(256) void k_cvt(const float* __restrict__ wq,
                                             const float* __restrict__ wk,
                                             const float* __restrict__ wv,
                                             const float* __restrict__ wg,
                                             const float* __restrict__ wo,
                                             _Float16* __restrict__ w16) {
    int idx = blockIdx.x * 256 + threadIdx.x;      // 5*16384 = 81920 total
    int m = idx >> 14;
    int r = idx & 16383;
    const float* src = (m == 0) ? wq : (m == 1) ? wk : (m == 2) ? wv
                     : (m == 3) ? wg : wo;
    float scl = (m == 0) ? 0.17677669529663688f /* 1/sqrt(32) */ : 1.0f;
    w16[idx] = (_Float16)(src[r] * scl);
}

// ---------------- kernel 1: LayerNorm + triangle bias ----------------
// one wave32 per (i,j) position; lane owns 4 channels
__global__ __launch_bounds__(256) void k_ln(const float* __restrict__ x,
                                            const float* __restrict__ ln_g,
                                            const float* __restrict__ ln_b,
                                            const float* __restrict__ w_tri,
                                            _Float16* __restrict__ xn16,
                                            float* __restrict__ tri) {
    int wave = threadIdx.x >> 5;
    int lane = threadIdx.x & 31;
    int pos  = blockIdx.x * 8 + wave;              // 0 .. 65535
    const float* xp = x + (size_t)pos * CZ + lane * 4;
    v4f xv = *(const v4f*)xp;

    float s  = xv[0] + xv[1] + xv[2] + xv[3];
    float s2 = xv[0]*xv[0] + xv[1]*xv[1] + xv[2]*xv[2] + xv[3]*xv[3];
    s  = wave_sum(s);
    s2 = wave_sum(s2);
    float mu  = s * (1.0f / CZ);
    float var = s2 * (1.0f / CZ) - mu * mu;
    float rs  = rsqrtf(var + 1e-5f);

    float xn[4];
    v4h o16;
#pragma unroll
    for (int j = 0; j < 4; ++j) {
        int c = lane * 4 + j;
        xn[j] = (xv[j] - mu) * rs * ln_g[c] + ln_b[c];
        o16[j] = (_Float16)xn[j];
    }
    *(v4h*)(xn16 + (size_t)pos * CZ + lane * 4) = o16;

#pragma unroll
    for (int h = 0; h < NH; ++h) {
        float p = 0.f;
#pragma unroll
        for (int j = 0; j < 4; ++j) p += xn[j] * w_tri[h * CZ + lane * 4 + j];
        p = wave_sum(p);
        if (lane == 0) tri[(size_t)h * NN * NN + pos] = p;
    }
}

// ---------------- kernel 2: Q/K/V/G projections (WMMA) ----------------
// grid.x = 65536/16 row tiles, grid.y = matrix {q,k,v,g}; 8 waves = 8 col tiles
// V is written TRANSPOSED per (i,h): vtg layout [i][h][d][k] -> each (i,h) block
// is a flat contiguous 16 KB region, perfect for a 1-descriptor TDM copy.
__global__ __launch_bounds__(256) void k_proj(const _Float16* __restrict__ xn16,
                                              const _Float16* __restrict__ w16,
                                              const float* __restrict__ bg,
                                              _Float16* __restrict__ q16,
                                              _Float16* __restrict__ k16,
                                              _Float16* __restrict__ vtg,
                                              _Float16* __restrict__ g16) {
    __shared__ _Float16 atile[16 * CZ];            // 4 KB
    int m       = blockIdx.y;
    int rowbase = blockIdx.x * 16;
    int tid     = threadIdx.x;

    *(v8h*)(atile + tid * 8) = *(const v8h*)(xn16 + (size_t)rowbase * CZ + tid * 8);
    __syncthreads();

    int wave = tid >> 5, lane = tid & 31;
    int colbase = wave * 16;
    const _Float16* bt = w16 + (size_t)m * CZ * CZ;

    v8f c = {0.f, 0.f, 0.f, 0.f, 0.f, 0.f, 0.f, 0.f};
#pragma unroll
    for (int ks = 0; ks < 4; ++ks) {
        v16h a = load_a(atile + ks * 32, CZ, lane);
        v16h b = load_b(bt + (size_t)colbase * CZ + ks * 32, CZ, lane);
        c = wmma_f16(a, b, c);
    }

    int col = colbase + (lane & 15);
    int r0  = rowbase + ((lane < 16) ? 0 : 8);
    if (m == 2) {
        // V transposed store: [i][h][d][k]
        int hh = col >> 5, d = col & 31;
        int ii = rowbase >> 8;                     // 16-row tile never straddles i
#pragma unroll
        for (int e = 0; e < 8; ++e) {
            int kidx = (r0 + e) & 255;
            vtg[(((size_t)ii * NH + hh) * DH + d) * NN + kidx] = (_Float16)c[e];
        }
    } else {
        _Float16* dst = (m == 0) ? q16 : (m == 1) ? k16 : g16;
        float bgv = (m == 3) ? bg[col] : 0.f;
#pragma unroll
        for (int e = 0; e < 8; ++e) {
            float val = c[e];
            if (m == 3) val = 1.0f / (1.0f + __expf(-(val + bgv)));
            dst[(size_t)(r0 + e) * CZ + col] = (_Float16)val;
        }
    }
}

// ---------------- kernel 3: attention (QK^T + softmax + PV + gate) ----------
// block = 64 threads (2 waves); one block handles (i, h, 32 q-rows)
__global__ __launch_bounds__(64) void k_attn(const _Float16* __restrict__ q16,
                                             const _Float16* __restrict__ k16,
                                             const _Float16* __restrict__ vtg,
                                             const _Float16* __restrict__ g16,
                                             const float* __restrict__ tri,
                                             const float* __restrict__ mask,
                                             _Float16* __restrict__ o16) {
    __shared__ _Float16 vt[DH][NN + 8];            // Vt tile [d][k], padded rows
    __shared__ float    lg[2][16][NN];             // per-wave logits, 32 KB
                                                   // (f16 probs overwrite in place)
    int bid = blockIdx.x;                          // N * H * 8 = 8192
    int i   = bid >> 5;
    int h   = (bid >> 3) & 3;
    int qb  = (bid & 7) * 32;
    int tid = threadIdx.x, wave = tid >> 5, lane = tid & 31;

    // ---- stage contiguous 16 KB Vt block into LDS (TDM pads 16B per 512B row)
    const _Float16* vsrc = vtg + (size_t)(i * NH + h) * DH * NN;
#if HAVE_TDM
    if (tid < 32) {
        unsigned long long ga = (unsigned long long)vsrc;
        v4u g0;
        g0[0] = 1u;                                        // count = 1
        g0[1] = lds_addr_of(&vt[0][0]);                    // lds_addr
        g0[2] = (unsigned)(ga & 0xffffffffu);              // global_addr[31:0]
        g0[3] = (unsigned)((ga >> 32) & 0x01ffffffu)       // global_addr[56:32]
              | 0x80000000u;                               // type = 2 ("image")
        v8i g1;
        g1[0] = (1 << 16)          // data_size = 2 bytes
              | (1 << 20)          // pad_enable
              | (6 << 22)          // pad_interval: 128 DWORDs (one 512B row)
              | (3 << 25);         // pad_amount: 4 DWORDs (16 bytes)
        g1[1] = (int)(8192u << 16);                        // tensor_dim0 = 8192
        g1[2] = (int)(1u << 16);                           // tensor_dim1 = 1
        g1[3] = (int)(8192u << 16);                        // tile_dim0 = 8192
        g1[4] = 1;                                         // tile_dim1 = 1
        g1[5] = 8192;                                      // tensor_dim0_stride
        g1[6] = 0;
        g1[7] = 0;
        v4i g2 = {0, 0, 0, 0};
        v4i g3 = {0, 0, 0, 0};
        v8i g4 = {0, 0, 0, 0, 0, 0, 0, 0};                 // unused extra group
        __builtin_amdgcn_tensor_load_to_lds(g0, g1, g2, g3, g4, 0);
        __builtin_amdgcn_s_wait_tensorcnt(0);
    }
#else
    for (int it = 0; it < 16; ++it) {
        int f   = (tid + it * 64) * 8;             // half index 0..8191
        int row = f >> 8;
        int ck  = f & 255;
        *(v8h*)&vt[row][ck] = *(const v8h*)(vsrc + f);
    }
#endif
    __syncthreads();

    int qbase = qb + wave * 16;

    // ---- logits = Q K^T  (K-dim = DH = 32 -> one WMMA per 16x16 tile) ----
    const _Float16* qp = q16 + ((size_t)(i * NN + qbase)) * CZ + h * DH;
    v16h a = load_a(qp, CZ, lane);
#pragma unroll
    for (int kt = 0; kt < 16; ++kt) {
        const _Float16* kp = k16 + ((size_t)(i * NN + kt * 16)) * CZ + h * DH;
        v16h b = load_b(kp, CZ, lane);
        v8f c = {0.f, 0.f, 0.f, 0.f, 0.f, 0.f, 0.f, 0.f};
        c = wmma_f16(a, b, c);
        int col = kt * 16 + (lane & 15);
        int rr  = (lane < 16) ? 0 : 8;
#pragma unroll
        for (int e = 0; e < 8; ++e) lg[wave][rr + e][col] = c[e];
    }

    // ---- biases + softmax (wave-local; probs written in place as f16) ----
    const float* trih = tri + (size_t)h * NN * NN;
    for (int r = 0; r < 16; ++r) {
        int qrow = qbase + r;
        float vals[8];
        float mx = -1e30f;
#pragma unroll
        for (int j = 0; j < 8; ++j) {
            int cc = lane * 8 + j;
            float t = lg[wave][r][cc] + trih[(size_t)qrow * NN + cc]
                    + INFB * (mask[(size_t)i * NN + cc] - 1.0f);
            vals[j] = t;
            mx = fmaxf(mx, t);
        }
        mx = wave_max(mx);
        float sum = 0.f;
#pragma unroll
        for (int j = 0; j < 8; ++j) { vals[j] = __expf(vals[j] - mx); sum += vals[j]; }
        sum = wave_sum(sum);
        float inv = 1.0f / sum;
        v8h p8;
#pragma unroll
        for (int j = 0; j < 8; ++j) p8[j] = (_Float16)(vals[j] * inv);
        _Float16* prow = (_Float16*)&lg[wave][r][0];   // reuse logits row storage
        *(v8h*)(prow + lane * 8) = p8;
    }

    // ---- O = P V  (probs f16 at row stride 512 halfs), gate, store --------
    const _Float16* pbase = (const _Float16*)&lg[wave][0][0];
#pragma unroll
    for (int dt = 0; dt < 2; ++dt) {
        v8f c = {0.f, 0.f, 0.f, 0.f, 0.f, 0.f, 0.f, 0.f};
#pragma unroll
        for (int ks = 0; ks < 8; ++ks) {
            v16h aa = load_a(pbase + ks * 32, 2 * NN, lane);   // ld = 512 halfs
            v16h bb = load_b(&vt[dt * 16][ks * 32], NN + 8, lane);
            c = wmma_f16(aa, bb, c);
        }
        int col = dt * 16 + (lane & 15);
        int r0  = qbase + ((lane < 16) ? 0 : 8);
#pragma unroll
        for (int e = 0; e < 8; ++e) {
            size_t oi = ((size_t)(i * NN + r0 + e)) * CZ + h * DH + col;
            float gv = (float)g16[oi];
            o16[oi] = (_Float16)(c[e] * gv);
        }
    }
}

// ---------------- kernel 4: output projection (WMMA) + bias ----------------
__global__ __launch_bounds__(256) void k_out(const _Float16* __restrict__ o16,
                                             const _Float16* __restrict__ wo16,
                                             const float* __restrict__ bo,
                                             float* __restrict__ out) {
    __shared__ _Float16 atile[16 * CZ];
    int rowbase = blockIdx.x * 16;
    int tid = threadIdx.x;
    *(v8h*)(atile + tid * 8) = *(const v8h*)(o16 + (size_t)rowbase * CZ + tid * 8);
    __syncthreads();

    int wave = tid >> 5, lane = tid & 31;
    int colbase = wave * 16;
    v8f c = {0.f, 0.f, 0.f, 0.f, 0.f, 0.f, 0.f, 0.f};
#pragma unroll
    for (int ks = 0; ks < 4; ++ks) {
        v16h a = load_a(atile + ks * 32, CZ, lane);
        v16h b = load_b(wo16 + (size_t)colbase * CZ + ks * 32, CZ, lane);
        c = wmma_f16(a, b, c);
    }
    int col = colbase + (lane & 15);
    int r0  = rowbase + ((lane < 16) ? 0 : 8);
    float bv = bo[col];
#pragma unroll
    for (int e = 0; e < 8; ++e)
        out[(size_t)(r0 + e) * CZ + col] = c[e] + bv;
}

// ---------------- launcher ----------------
extern "C" void kernel_launch(void* const* d_in, const int* in_sizes, int n_in,
                              void* d_out, int out_size, void* d_ws, size_t ws_size,
                              hipStream_t stream) {
    const float* x     = (const float*)d_in[0];
    const float* mask  = (const float*)d_in[1];
    const float* ln_g  = (const float*)d_in[2];
    const float* ln_b  = (const float*)d_in[3];
    const float* w_tri = (const float*)d_in[4];
    const float* wq    = (const float*)d_in[5];
    const float* wk    = (const float*)d_in[6];
    const float* wv    = (const float*)d_in[7];
    const float* wg    = (const float*)d_in[8];
    const float* bg    = (const float*)d_in[9];
    const float* wo    = (const float*)d_in[10];
    const float* bo    = (const float*)d_in[11];

    char* ws = (char*)d_ws;
    size_t off = 0;
    auto take = [&](size_t bytes) {
        size_t o = off;
        off += (bytes + 255) & ~(size_t)255;
        return o;
    };
    const size_t posCZh = (size_t)NN * NN * CZ * sizeof(_Float16);   // 16.78 MB
    _Float16* xn16 = (_Float16*)(ws + take(posCZh));
    _Float16* w16  = (_Float16*)(ws + take((size_t)5 * CZ * CZ * sizeof(_Float16)));
    float*    tri  = (float*)   (ws + take((size_t)NH * NN * NN * sizeof(float)));
    _Float16* q16  = (_Float16*)(ws + take(posCZh));
    _Float16* k16  = (_Float16*)(ws + take(posCZh));
    _Float16* vtg  = (_Float16*)(ws + take(posCZh));
    _Float16* g16  = (_Float16*)(ws + take(posCZh));
    _Float16* o16  = (_Float16*)(ws + take(posCZh));

    k_cvt<<<(5 * CZ * CZ) / 256, 256, 0, stream>>>(wq, wk, wv, wg, wo, w16);
    k_ln<<<(NN * NN) / 8, 256, 0, stream>>>(x, ln_g, ln_b, w_tri, xn16, tri);
    k_proj<<<dim3((NN * NN) / 16, 4), 256, 0, stream>>>(xn16, w16, bg,
                                                        q16, k16, vtg, g16);
    k_attn<<<NN * NH * 8, 64, 0, stream>>>(q16, k16, vtg, g16, tri, mask, o16);
    k_out<<<(NN * NN) / 16, 256, 0, stream>>>(o16, w16 + (size_t)4 * CZ * CZ,
                                              bo, (float*)d_out);
}
